// Net_SLSTM_26517128086133
// MI455X (gfx1250) — 8-bit matrix operand order repaired and checked by reading, compiled, not run on hardware
//
#include <hip/hip_runtime.h>
#include <math.h>

#define T_STEPS 1024
#define BATCH   256
#define CIN     14
#define HID     128
#define G4      512
#define NCLS    7
#define BW      16
#define NBLK    (BATCH / BW)
#define NTHREADS 256

#define OFF_W     0
#define OFF_GATES (OFF_W + 4 * 32 * 2048)
#define OFF_XA    (OFF_GATES + 32768)
#define OFF_M1A   (OFF_XA + 2048)
#define OFF_S1A   (OFF_M1A + 2048)
#define OFF_M2A   (OFF_S1A + 2048)
#define OFF_B1    (OFF_M2A + 2048)
#define OFF_B2    (OFF_B1 + 2048)
#define SMEM_BYTES (OFF_B2 + 2048)

typedef int   v16i __attribute__((ext_vector_type(16)));
typedef float v8f  __attribute__((ext_vector_type(8)));

#if defined(__has_builtin)
#  if __has_builtin(__builtin_amdgcn_cvt_pk_fp8_f32)
#    define HAVE_HW_FP8 1
#  endif
#endif

__device__ __forceinline__ unsigned char f32_to_e4m3_sw(float x) {
  unsigned u = __float_as_uint(x);
  unsigned s = (u >> 24) & 0x80u;
  float a = fabsf(x);
  if (!(a >= 0x1p-9f))  return (unsigned char)s;
  if (a >= 448.0f)      return (unsigned char)(s | 0x7Eu);
  int e = (int)((u >> 23) & 0xFF) - 127;
  unsigned mr = ((u & 0x7FFFFFu) + 0x80000u) >> 20;
  if (mr == 8u) { mr = 0u; e += 1; }
  int E = e + 7;
  if (E <= 0) {
    int dm = (int)(a * 512.0f + 0.5f);
    if (dm > 7) dm = 7;
    return (unsigned char)(s | (unsigned)dm);
  }
  if (E >= 16 || (E == 15 && mr == 7u)) return (unsigned char)(s | 0x7Eu);
  return (unsigned char)(s | ((unsigned)E << 3) | mr);
}

__device__ __forceinline__ unsigned char cvt_fp8(float x) {
#ifdef HAVE_HW_FP8
  return (unsigned char)(__builtin_amdgcn_cvt_pk_fp8_f32(x, x, 0, false) & 0xFF);
#else
  return f32_to_e4m3_sw(x);
#endif
}

__device__ __forceinline__ int a_frag_off(int m, int k) {
  int lane = (m & 15) | (((k >> 3) & 1) << 4);
  int kk = k & 63;
  int v = ((((kk >> 4) & 3) << 1) | ((kk >> 2) & 1));
  int byteIdx = ((k >> 6) << 5) + (v << 2) + (kk & 3);
  return lane * 64 + byteIdx;
}
__device__ __forceinline__ int b_frag_off(int k, int n) {
  int lane = (n & 15) | (((k >> 3) & 1) << 4);
  int byteIdx = ((k >> 4) << 3) + (k & 7);
  return lane * 64 + byteIdx;
}

__device__ __forceinline__ float sigf(float x) {
  return __builtin_amdgcn_rcpf(1.0f + __expf(-x));
}
__device__ __forceinline__ float tanh_fast(float x) {
  return 2.0f * sigf(2.0f * x) - 1.0f;
}

__global__ __launch_bounds__(NTHREADS)
void slstm_persistent_kernel(const float* __restrict__ x,
                             const float* __restrict__ Wih1, const float* __restrict__ Whh1,
                             const float* __restrict__ bih1, const float* __restrict__ bhh1,
                             const float* __restrict__ thr1p,
                             const float* __restrict__ Wih2, const float* __restrict__ Whh2,
                             const float* __restrict__ bih2, const float* __restrict__ bhh2,
                             const float* __restrict__ thr2p,
                             const float* __restrict__ fcw, const float* __restrict__ fcb,
                             float* __restrict__ out) {
  extern __shared__ unsigned char smem[];
  const int tid   = (int)threadIdx.x;
  const int lane  = tid & 31;
  const int wv    = tid >> 5;
  const int bbase = (int)blockIdx.x * BW;

  for (int i = tid * 4; i < 4 * 2048; i += NTHREADS * 4)
    *(unsigned*)(smem + OFF_XA + i) = 0u;

  for (int n = tid; n < G4; n += NTHREADS) {
    ((float*)(smem + OFF_B1))[n] = bih1[n] + bhh1[n];
    ((float*)(smem + OFF_B2))[n] = bih2[n] + bhh2[n];
  }

  for (int idx = tid; idx < HID * G4; idx += NTHREADS) {
    int k = idx >> 9;
    int n = idx & 511;
    int nt = n >> 4, nn = n & 15;
    int fo = b_frag_off(k, nn);
    float w0 = (k < CIN) ? Wih1[n * CIN + k] : 0.0f;
    smem[OFF_W + (size_t)(0 * 32 + nt) * 2048 + fo] = cvt_fp8(w0);
    smem[OFF_W + (size_t)(1 * 32 + nt) * 2048 + fo] = cvt_fp8(Whh1[n * HID + k]);
    smem[OFF_W + (size_t)(2 * 32 + nt) * 2048 + fo] = cvt_fp8(Wih2[n * HID + k]);
    smem[OFF_W + (size_t)(3 * 32 + nt) * 2048 + fo] = cvt_fp8(Whh2[n * HID + k]);
  }

  const float thr1 = thr1p[0];
  const float thr2 = thr2p[0];

  float syn1[8], mem1v[8], syn2[8], mem2v[8], msum[8];
#pragma unroll
  for (int j = 0; j < 8; ++j) { syn1[j] = mem1v[j] = syn2[j] = mem2v[j] = msum[j] = 0.0f; }

  float* gates = (float*)(smem + OFF_GATES);
  const float* bs1 = (const float*)(smem + OFF_B1);
  const float* bs2 = (const float*)(smem + OFF_B2);

  for (int t = 0; t < T_STEPS; ++t) {
    if (tid < BW * CIN) {
      int b = tid / CIN, c = tid % CIN;
      float xv = x[((size_t)t * BATCH + bbase + b) * CIN + c];
      smem[OFF_XA + a_frag_off(b, c)] = cvt_fp8(xv);
    }
    __syncthreads();

    {
      v16i a0 = *(const v16i*)(smem + OFF_XA  + lane * 64);
      v16i a1 = *(const v16i*)(smem + OFF_M1A + lane * 64);
#pragma unroll
      for (int ti = 0; ti < 4; ++ti) {
        int nt = wv * 4 + ti;
        v16i b0 = *(const v16i*)(smem + OFF_W + (size_t)(0 * 32 + nt) * 2048 + lane * 64);
        v16i b1 = *(const v16i*)(smem + OFF_W + (size_t)(1 * 32 + nt) * 2048 + lane * 64);
        v8f acc = {};
        acc = __builtin_amdgcn_wmma_f32_16x16x128_fp8_fp8(a0, b0, (short)0, acc, false, false);
        acc = __builtin_amdgcn_wmma_f32_16x16x128_fp8_fp8(a1, b1, (short)0, acc, false, false);
        int m0 = (lane >> 4) << 3;
        int n  = nt * 16 + (lane & 15);
#pragma unroll
        for (int r = 0; r < 8; ++r) gates[(m0 + r) * G4 + n] = acc[r];
      }
    }
    __syncthreads();

#pragma unroll
    for (int j = 0; j < 8; ++j) {
      int e = tid + NTHREADS * j;
      int b = e >> 7, h = e & 127;
      float gi = gates[b * G4 + h]       + bs1[h];
      float gf = gates[b * G4 + h + 128] + bs1[h + 128];
      float gg = gates[b * G4 + h + 256] + bs1[h + 256];
      float go = gates[b * G4 + h + 384] + bs1[h + 384];
      float sn = sigf(gf) * syn1[j] + sigf(gi) * tanh_fast(gg);
      float reset = (mem1v[j] > thr1) ? 1.0f : 0.0f;
      float mn = sigf(go) * tanh_fast(sn) - reset * thr1;
      syn1[j] = sn; mem1v[j] = mn;
      int fo = a_frag_off(b, h);
      smem[OFF_S1A + fo] = (mn > thr1) ? (unsigned char)0x38 : (unsigned char)0;
      smem[OFF_M1A + fo] = cvt_fp8(mn);
    }
    __syncthreads();

    {
      v16i a0 = *(const v16i*)(smem + OFF_S1A + lane * 64);
      v16i a1 = *(const v16i*)(smem + OFF_M2A + lane * 64);
#pragma unroll
      for (int ti = 0; ti < 4; ++ti) {
        int nt = wv * 4 + ti;
        v16i b0 = *(const v16i*)(smem + OFF_W + (size_t)(2 * 32 + nt) * 2048 + lane * 64);
        v16i b1 = *(const v16i*)(smem + OFF_W + (size_t)(3 * 32 + nt) * 2048 + lane * 64);
        v8f acc = {};
        acc = __builtin_amdgcn_wmma_f32_16x16x128_fp8_fp8(a0, b0, (short)0, acc, false, false);
        acc = __builtin_amdgcn_wmma_f32_16x16x128_fp8_fp8(a1, b1, (short)0, acc, false, false);
        int m0 = (lane >> 4) << 3;
        int n  = nt * 16 + (lane & 15);
#pragma unroll
        for (int r = 0; r < 8; ++r) gates[(m0 + r) * G4 + n] = acc[r];
      }
    }
    __syncthreads();

#pragma unroll
    for (int j = 0; j < 8; ++j) {
      int e = tid + NTHREADS * j;
      int b = e >> 7, h = e & 127;
      float gi = gates[b * G4 + h]       + bs2[h];
      float gf = gates[b * G4 + h + 128] + bs2[h + 128];
      float gg = gates[b * G4 + h + 256] + bs2[h + 256];
      float go = gates[b * G4 + h + 384] + bs2[h + 384];
      float sn = sigf(gf) * syn2[j] + sigf(gi) * tanh_fast(gg);
      float reset = (mem2v[j] > thr2) ? 1.0f : 0.0f;
      float mn = sigf(go) * tanh_fast(sn) - reset * thr2;
      syn2[j] = sn; mem2v[j] = mn;
      msum[j] += mn;
      smem[OFF_M2A + a_frag_off(b, h)] = cvt_fp8(mn);
    }

    if (t + 1 < T_STEPS && tid < BW) {
      __builtin_prefetch(&x[((size_t)(t + 1) * BATCH + bbase + tid) * CIN], 0, 1);
    }
  }

  __syncthreads();
  float* fin = gates;
#pragma unroll
  for (int j = 0; j < 8; ++j) {
    int e = tid + NTHREADS * j;
    fin[e] = msum[j] * (1.0f / (float)T_STEPS);
  }
  __syncthreads();
  if (tid < BW * NCLS) {
    int b = tid / NCLS, c = tid % NCLS;
    float acc = fcb[c];
#pragma unroll 4
    for (int h = 0; h < HID; ++h) acc += fin[b * HID + h] * fcw[c * HID + h];
    out[(size_t)(bbase + b) * NCLS + c] = acc;
  }
}

extern "C" void kernel_launch(void* const* d_in, const int* in_sizes, int n_in,
                              void* d_out, int out_size, void* d_ws, size_t ws_size,
                              hipStream_t stream) {
  (void)in_sizes; (void)n_in; (void)d_ws; (void)ws_size; (void)out_size;
  const float* x    = (const float*)d_in[0];
  const float* Wih1 = (const float*)d_in[1];
  const float* Whh1 = (const float*)d_in[2];
  const float* bih1 = (const float*)d_in[3];
  const float* bhh1 = (const float*)d_in[4];
  const float* thr1 = (const float*)d_in[5];
  const float* Wih2 = (const float*)d_in[6];
  const float* Whh2 = (const float*)d_in[7];
  const float* bih2 = (const float*)d_in[8];
  const float* bhh2 = (const float*)d_in[9];
  const float* thr2 = (const float*)d_in[10];
  const float* fcw  = (const float*)d_in[11];
  const float* fcb  = (const float*)d_in[12];

  dim3 grid(NBLK), block(NTHREADS);
  hipLaunchKernelGGL(slstm_persistent_kernel, grid, block, SMEM_BYTES, stream,
                     x, Wih1, Whh1, bih1, bhh1, thr1,
                     Wih2, Whh2, bih2, bhh2, thr2, fcw, fcb,
                     (float*)d_out);
}
